// LatentPositionHead_79336635892072
// MI455X (gfx1250) — matrix operand order repaired and checked by reading, not compiled, not run on hardware
//
#include <hip/hip_runtime.h>
#include <hip/hip_bf16.h>

// ---------------------------------------------------------------------------
// MI455X (gfx1250) implementation. wave32, WMMA f32<-f16 16x16x32 for every
// GEMM (pw1/pw2/bn1x1, stem 3x3 convs as implicit GEMM, MLP head).
// All WMMA operand fragments are loaded as contiguous 2x128-bit per lane.
// ---------------------------------------------------------------------------

typedef __attribute__((ext_vector_type(16))) _Float16 v16h;
typedef __attribute__((ext_vector_type(8)))  _Float16 v8h;
typedef __attribute__((ext_vector_type(8)))  float    v8f;

#define WMMA_F16(a, b, c) \
  __builtin_amdgcn_wmma_f32_16x16x32_f16(false, (a), false, (b), (short)0, (c), false, false)

union FragU { v16h v; v8h h[2]; };

__device__ __forceinline__ float gelu_exact(float x) {
  return 0.5f * x * (1.0f + erff(x * 0.70710678118654752f));
}

// A fragment (16x32 tile at m0,k0) from row-major W[M][K] f16.
// Per lane: M = m0 + lane%16, K = k0 + 16*half + i -> 16 contiguous halves.
__device__ __forceinline__ v16h load_a_frag(const _Float16* W, int m0, int k0, int ldk) {
  const int lane = threadIdx.x & 31;
  const int hf   = lane >> 4;
  const int m    = m0 + (lane & 15);
  const _Float16* base = W + (size_t)m * ldk + k0 + hf * 16;
  FragU u;
  u.h[0] = *(const v8h*)(base);
  u.h[1] = *(const v8h*)(base + 8);
  return u.v;
}

// B fragment (32x16 tile at k0,n0) from COLUMN-major Bc[N][K] f16 (ldk = K).
// Per lane: N = n0 + lane%16, K = k0 + 16*half + i -> 16 contiguous halves.
__device__ __forceinline__ v16h load_b_frag_cm(const _Float16* Bc, int k0, int n0, int ldk) {
  const int lane = threadIdx.x & 31;
  const int hf   = lane >> 4;
  const int n    = n0 + (lane & 15);
  const _Float16* base = Bc + (size_t)n * ldk + k0 + hf * 16;
  FragU u;
  u.h[0] = *(const v8h*)(base);
  u.h[1] = *(const v8h*)(base + 8);
  return u.v;
}

// Implicit-GEMM B fragment for 3x3 conv over a 17x17 tile stored pixel-major
// with a 1-px zero halo: win2[19][19][64] f16.  K is reordered as
// k = (dy*3+dx)*64 + c  (576 total), so each lane's 16 elements are 16
// contiguous channels at a single tap -> one 32-byte contiguous LDS load.
__device__ __forceinline__ v16h load_b_im2col(const _Float16* win2, int k0, int n0) {
  const int lane = threadIdx.x & 31;
  const int hf   = lane >> 4;
  int p = n0 + (lane & 15);
  if (p > 288) p = 288;                 // clamp garbage columns (discarded at store)
  const int py = p / 17, px = p - py * 17;
  const int k  = k0 + hf * 16;          // multiple of 16, stays within one tap
  const int r  = k >> 6, c0 = k & 63;
  const int dy = r / 3, dx = r - dy * 3;
  const _Float16* base = win2 + ((py + dy) * 19 + (px + dx)) * 64 + c0;
  FragU u;
  u.h[0] = *(const v8h*)(base);
  u.h[1] = *(const v8h*)(base + 8);
  return u.v;
}

__device__ __forceinline__ float sample_bilinear(const float* plane, int H, int W,
                                                 float xs, float ys) {
  float x0f = floorf(xs), y0f = floorf(ys);
  int   x0  = (int)x0f,  y0  = (int)y0f;
  float fx  = xs - x0f,  fy  = ys - y0f;
  float acc = 0.f;
#pragma unroll
  for (int t = 0; t < 4; ++t) {
    int   yi = y0 + (t >> 1), xi = x0 + (t & 1);
    float w  = ((t >> 1) ? fy : 1.f - fy) * ((t & 1) ? fx : 1.f - fx);
    if (yi >= 0 && yi < H && xi >= 0 && xi < W) acc += plane[yi * W + xi] * w;
  }
  return acc;
}

// ---------------- workspace layout (element offsets within f16 region) -----
static constexpr size_t OFF_W1B   = 0;                          // 1024x256   (A, row-major)
static constexpr size_t OFF_W2B   = OFF_W1B  + 1024ull * 256;   // 256x1024   (A, row-major)
static constexpr size_t OFF_BNW   = OFF_W2B  + 256ull * 1024;   // 128x256    (A, row-major)
static constexpr size_t OFF_ST1W  = OFF_BNW  + 128ull * 256;    // 64x576, K reordered tap-major
static constexpr size_t OFF_ST2W  = OFF_ST1W + 64ull * 576;     // 64x576, K reordered tap-major
static constexpr size_t OFF_H1W   = OFF_ST2W + 64ull * 576;     // 128x192    (B col-major = as-is)
static constexpr size_t OFF_H2W   = OFF_H1W  + 128ull * 192;    // 128x128    (B col-major = as-is)
static constexpr size_t F16_TOTAL = OFF_H2W  + 128ull * 128;

// ---------------------------------------------------------------------------
__global__ __launch_bounds__(256)
void prep_weights_kernel(const float* __restrict__ pw1_w, const float* __restrict__ pw2_w,
                         const float* __restrict__ bn1x1_w,
                         const float* __restrict__ st1_w, const float* __restrict__ st2_w,
                         const float* __restrict__ h1_w, const float* __restrict__ h2_w,
                         _Float16* __restrict__ wsh) {
  int i = blockIdx.x * blockDim.x + threadIdx.x;
  const int n1 = 1024 * 256, n2 = 256 * 1024, n3 = 128 * 256;
  const int n4 = 64 * 576, n5 = 64 * 576, n6 = 128 * 192, n7 = 128 * 128;
  if (i < n1) { wsh[OFF_W1B + i] = (_Float16)pw1_w[i]; return; } i -= n1;
  if (i < n2) { wsh[OFF_W2B + i] = (_Float16)pw2_w[i]; return; } i -= n2;
  if (i < n3) { wsh[OFF_BNW + i] = (_Float16)bn1x1_w[i]; return; } i -= n3;
  if (i < n4) { // dst k-order: k = tap*64 + c ; src: o*576 + c*9 + tap
    int o = i / 576, kd = i - o * 576;
    int r = kd >> 6, c = kd & 63;
    wsh[OFF_ST1W + i] = (_Float16)st1_w[o * 576 + c * 9 + r]; return;
  } i -= n4;
  if (i < n5) {
    int o = i / 576, kd = i - o * 576;
    int r = kd >> 6, c = kd & 63;
    wsh[OFF_ST2W + i] = (_Float16)st2_w[o * 576 + c * 9 + r]; return;
  } i -= n5;
  if (i < n6) { wsh[OFF_H1W + i] = (_Float16)h1_w[i]; return; } i -= n6;
  if (i < n7) { wsh[OFF_H2W + i] = (_Float16)h2_w[i]; }
}

// ---------------------------------------------------------------------------
// Bottleneck branch: 1 block (8 waves) per source.
// Activations staged COLUMN-major: [pixel(32)][features] for fast B frags.
struct BnPhaseA {
  float    win[256 * 25];           // bilinear window (residual), f32
  _Float16 xln[32 * 256];           // [p][c] dw-out then LN-out
  _Float16 y1 [32 * 128];           // [p][hidden-chunk] pw1+GELU
};
struct BnPhaseB {
  float    win[256 * 25];           // same bytes as PhaseA.win (persists)
  _Float16 x2 [32 * 256];           // [p][c] residual output
  float    zred[128 * 25];
};
union BnSmem { BnPhaseA a; BnPhaseB b; };

__global__ __launch_bounds__(256)
void bn_branch_kernel(const float* __restrict__ bottleneck, const float* __restrict__ pos,
                      const int* __restrict__ fh, const int* __restrict__ fw,
                      const int* __restrict__ vh, const int* __restrict__ vw,
                      const float* __restrict__ dw_w, const float* __restrict__ dw_b,
                      const float* __restrict__ ln_g, const float* __restrict__ ln_b,
                      const float* __restrict__ pw1_b, const float* __restrict__ pw2_b,
                      const float* __restrict__ cn_gamma, const float* __restrict__ bn1x1_b,
                      const _Float16* __restrict__ w1b, const _Float16* __restrict__ w2b,
                      const _Float16* __restrict__ bnwb, float* __restrict__ bn_vec) {
  __shared__ BnSmem sm;
  __shared__ float mu_s[25], rs_s[25], gw_s[25];
  const int n = blockIdx.x, tid = threadIdx.x;
  const int lane = tid & 31, wave = tid >> 5;

  const float sx = (float)fw[0] / (float)vw[0];
  const float sy = (float)fh[0] / (float)vh[0];
  const float px0 = pos[n * 2 + 0] * sx;
  const float py0 = pos[n * 2 + 1] * sy;

  if (tid == 0) {                    // 5x5 gaussian weights
    float s = 0.f;
    for (int p = 0; p < 25; ++p) {
      float ty = -1.f + 0.5f * (float)(p / 5);
      float tx = -1.f + 0.5f * (float)(p % 5);
      float g = expf(-(tx * tx + ty * ty) / 0.32f);
      gw_s[p] = g; s += g;
    }
    for (int p = 0; p < 25; ++p) gw_s[p] /= s;
  }

  { // bilinear 5x5 window, one channel per thread
    const int c = tid;
    const float* plane = bottleneck + (size_t)c * (128 * 128);
    for (int p = 0; p < 25; ++p) {
      float xs = px0 + (float)(p % 5) - 2.f;
      float ys = py0 + (float)(p / 5) - 2.f;
      sm.a.win[c * 25 + p] = sample_bilinear(plane, 128, 128, xs, ys);
    }
  }
  // zero pad rows p=25..31 of xln
  for (int i = tid; i < 7 * 256; i += 256) sm.a.xln[25 * 256 + i] = (_Float16)0.f;
  __syncthreads();

  { // depthwise 7x7, zero padding 3 (window-local); write transposed [p][c]
    const int c = tid;
    const float bia = dw_b[c];
    for (int p = 0; p < 25; ++p) {
      int j = p / 5, i = p - j * 5;
      float acc = bia;
      for (int dy = 0; dy < 7; ++dy) {
        int iy = j + dy - 3;
        if (iy < 0 || iy >= 5) continue;
        for (int dx = 0; dx < 7; ++dx) {
          int ix = i + dx - 3;
          if (ix < 0 || ix >= 5) continue;
          acc += sm.a.win[c * 25 + iy * 5 + ix] * dw_w[c * 49 + dy * 7 + dx];
        }
      }
      sm.a.xln[p * 256 + c] = (_Float16)acc;
    }
  }
  __syncthreads();

  if (tid < 25) {                    // LN stats: contiguous row read per pixel
    float s = 0.f, s2 = 0.f;
    for (int c = 0; c < 256; ++c) {
      float v = (float)sm.a.xln[tid * 256 + c];
      s += v; s2 += v * v;
    }
    float m = s * (1.f / 256.f);
    float var = s2 * (1.f / 256.f) - m * m;
    mu_s[tid] = m;
    rs_s[tid] = rsqrtf(var + 1e-6f);
  }
  __syncthreads();
  { // apply LN affine
    const int c = tid;
    const float g = ln_g[c], b = ln_b[c];
    for (int p = 0; p < 25; ++p) {
      float v = (float)sm.a.xln[p * 256 + c];
      sm.a.xln[p * 256 + c] = (_Float16)(((v - mu_s[p]) * rs_s[p]) * g + b);
    }
  }
  __syncthreads();

  // pw1 (256->1024) -> GELU -> pw2 (1024->256), fused over 8 hidden chunks of
  // 128; pw2 accumulators persist in registers (4 tiles per wave).
  v8f acc2[4] = {};
  for (int hc = 0; hc < 8; ++hc) {
#pragma unroll
    for (int t = 0; t < 2; ++t) {    // pw1: 16 tiles (8Mx2N) over 8 waves
      int tile = wave * 2 + t;
      int mt = tile >> 1, nt = tile & 1;
      v8f acc = {};
#pragma unroll
      for (int kk = 0; kk < 8; ++kk)
        acc = WMMA_F16(load_a_frag(w1b + (size_t)(hc * 128) * 256, mt * 16, kk * 32, 256),
                       load_b_frag_cm(sm.a.xln, kk * 32, nt * 16, 256), acc);
      const int ncol = nt * 16 + (lane & 15);
      const int mbase = mt * 16 + 8 * (lane >> 4);
#pragma unroll
      for (int v = 0; v < 8; ++v) {   // contiguous 8-half run per lane
        int mrow = mbase + v;
        sm.a.y1[ncol * 128 + mrow] = (_Float16)gelu_exact(acc[v] + pw1_b[hc * 128 + mrow]);
      }
    }
    __syncthreads();
#pragma unroll
    for (int t = 0; t < 4; ++t) {    // pw2 partial: 32 tiles (16Mx2N) over 8 waves
      int tile = wave * 4 + t;
      int mt = tile >> 1, nt = tile & 1;
#pragma unroll
      for (int kk = 0; kk < 4; ++kk)
        acc2[t] = WMMA_F16(load_a_frag(w2b, mt * 16, hc * 128 + kk * 32, 1024),
                           load_b_frag_cm(sm.a.y1, kk * 32, nt * 16, 128), acc2[t]);
    }
    __syncthreads();
  }

#pragma unroll
  for (int t = 0; t < 4; ++t) {      // residual + layer-scale -> x2 [p][c]
    int tile = wave * 4 + t;
    int mt = tile >> 1, nt = tile & 1;
    const int ncol = nt * 16 + (lane & 15);
    const int mbase = mt * 16 + 8 * (lane >> 4);
#pragma unroll
    for (int v = 0; v < 8; ++v) {
      int c = mbase + v;
      float val = acc2[t][v] + pw2_b[c];
      float r = (ncol < 25) ? sm.b.win[c * 25 + ncol] : 0.f;
      float xv = r + cn_gamma[c] * val;
      sm.b.x2[ncol * 256 + c] = (_Float16)((ncol < 25) ? xv : 0.f);
    }
  }
  __syncthreads();

#pragma unroll
  for (int t = 0; t < 2; ++t) {      // bn1x1: 16 tiles (8Mx2N)
    int tile = wave * 2 + t;
    int mt = tile >> 1, nt = tile & 1;
    v8f acc = {};
#pragma unroll
    for (int kk = 0; kk < 8; ++kk)
      acc = WMMA_F16(load_a_frag(bnwb, mt * 16, kk * 32, 256),
                     load_b_frag_cm(sm.b.x2, kk * 32, nt * 16, 256), acc);
    const int ncol = nt * 16 + (lane & 15);
    const int mbase = mt * 16 + 8 * (lane >> 4);
    if (ncol < 25) {
#pragma unroll
      for (int v = 0; v < 8; ++v) {
        int o = mbase + v;
        sm.b.zred[o * 25 + ncol] = gelu_exact(acc[v] + bn1x1_b[o]);
      }
    }
  }
  __syncthreads();

  if (tid < 128) {                   // gaussian-weighted pool
    float s = 0.f;
    for (int p = 0; p < 25; ++p) s += sm.b.zred[tid * 25 + p] * gw_s[p];
    bn_vec[(size_t)n * 128 + tid] = s;
  }
}

// ---------------------------------------------------------------------------
// Stem branch stage 1: sample 17x17x64 window (pixel-major, halo'd), st1 3x3
// conv as implicit GEMM; stage output as [n][289][64] f16 (contiguous stores).
__global__ __launch_bounds__(256)
void stem1_kernel(const float* __restrict__ vis, const float* __restrict__ pos,
                  const float* __restrict__ st1_b, const _Float16* __restrict__ st1wb,
                  _Float16* __restrict__ stage) {
  __shared__ _Float16 win2[361 * 64];                  // [19][19][64]
  const int n = blockIdx.x, tid = threadIdx.x;
  const int lane = tid & 31, wave = tid >> 5;
  const float px0 = pos[n * 2 + 0], py0 = pos[n * 2 + 1];

  for (int i = tid; i < 361 * 64; i += 256) win2[i] = (_Float16)0.f;
  __syncthreads();
  for (int i = tid; i < 64 * 289; i += 256) {
    int c = i / 289, p = i - c * 289;
    int py = p / 17, px = p - py * 17;
    float xs = px0 + (float)px - 8.f;
    float ys = py0 + (float)py - 8.f;
    win2[((py + 1) * 19 + (px + 1)) * 64 + c] =
        (_Float16)sample_bilinear(vis + (size_t)c * (1024 * 1024), 1024, 1024, xs, ys);
  }
  __syncthreads();

  _Float16* out = stage + (size_t)n * (289 * 64);      // [p][c]
  for (int t = wave; t < 76; t += 8) {   // 4 Mtiles x 19 Ntiles, K=576 (18 steps)
    int mt = t / 19, nt = t - mt * 19;
    v8f acc = {};
#pragma unroll
    for (int kk = 0; kk < 18; ++kk)
      acc = WMMA_F16(load_a_frag(st1wb, mt * 16, kk * 32, 576),
                     load_b_im2col(win2, kk * 32, nt * 16), acc);
    const int p = nt * 16 + (lane & 15);
    const int mbase = mt * 16 + 8 * (lane >> 4);
    if (p < 289) {
#pragma unroll
      for (int v = 0; v < 8; ++v) {     // 8 contiguous channels per lane
        int o = mbase + v;
        out[p * 64 + o] = (_Float16)gelu_exact(acc[v] + st1_b[o]);
      }
    }
  }
}

// Stem stage 2: st2 3x3 conv (implicit GEMM) + gaussian pool.
__global__ __launch_bounds__(256)
void stem2_kernel(const _Float16* __restrict__ stage, const float* __restrict__ st2_b,
                  const _Float16* __restrict__ st2wb, float* __restrict__ st_vec) {
  __shared__ _Float16 win2[361 * 64];
  __shared__ float gw[289];
  __shared__ float part[64];
  __shared__ float ginv;
  const int n = blockIdx.x, tid = threadIdx.x;
  const int lane = tid & 31, wave = tid >> 5;

  for (int i = tid; i < 361 * 64; i += 256) win2[i] = (_Float16)0.f;
  for (int i = tid; i < 289; i += 256) {
    float ty = -1.f + 0.125f * (float)(i / 17);
    float tx = -1.f + 0.125f * (float)(i % 17);
    gw[i] = expf(-(tx * tx + ty * ty) / 0.32f);
  }
  if (tid < 64) part[tid] = 0.f;
  __syncthreads();
  if (tid == 0) {
    float s = 0.f;
    for (int p = 0; p < 289; ++p) s += gw[p];
    ginv = 1.f / s;
  }
  // vectorized interior copy: 4 halves (8B) per iteration
  const _Float16* src = stage + (size_t)n * (289 * 64);
  for (int i = tid; i < 289 * 16; i += 256) {
    int p = i >> 4, q = (i & 15) * 4;
    int py = p / 17, px = p - py * 17;
    *(uint2*)(win2 + ((py + 1) * 19 + (px + 1)) * 64 + q) = *(const uint2*)(src + p * 64 + q);
  }
  __syncthreads();

  for (int t = wave; t < 76; t += 8) {
    int mt = t / 19, nt = t - mt * 19;
    v8f acc = {};
#pragma unroll
    for (int kk = 0; kk < 18; ++kk)
      acc = WMMA_F16(load_a_frag(st2wb, mt * 16, kk * 32, 576),
                     load_b_im2col(win2, kk * 32, nt * 16), acc);
    const int p = nt * 16 + (lane & 15);
    const int mbase = mt * 16 + 8 * (lane >> 4);
    if (p < 289) {
      const float w = gw[p] * ginv;
#pragma unroll
      for (int v = 0; v < 8; ++v) {
        int o = mbase + v;
        atomicAdd(&part[o], gelu_exact(acc[v] + st2_b[o]) * w);
      }
    }
  }
  __syncthreads();
  if (tid < 64) st_vec[(size_t)n * 64 + tid] = part[tid];
}

// ---------------------------------------------------------------------------
// Head MLP + sky transform: 64 rows per block (4 waves).
// Weights used as column-major B directly (h*_w[o][k]); A = activations (LDS).
__global__ __launch_bounds__(128)
void head_kernel(const float* __restrict__ bn_vec, const float* __restrict__ st_vec,
                 const float* __restrict__ p2s,
                 const float* __restrict__ h1_b, const float* __restrict__ h2_b,
                 const float* __restrict__ h3_w, const float* __restrict__ h3_b,
                 const _Float16* __restrict__ h1wh, const _Float16* __restrict__ h2wh,
                 float* __restrict__ out, int N) {
  __shared__ _Float16 xs[64 * 192];
  __shared__ _Float16 y1[64 * 128];
  __shared__ _Float16 y2[64 * 128];
  const int tid = threadIdx.x;
  const int lane = tid & 31, wave = tid >> 5;
  const int r0 = blockIdx.x * 64;

  for (int i = tid; i < 64 * 192; i += 128) {
    int lr = i / 192, k = i - lr * 192;
    int r = r0 + lr;
    float v = 0.f;
    if (r < N) v = (k < 128) ? bn_vec[(size_t)r * 128 + k] : st_vec[(size_t)r * 64 + (k - 128)];
    xs[i] = (_Float16)v;
  }
  __syncthreads();

  { // h1: M=64 (1 Mtile/wave), N=128 (8 tiles), K=192 (6 steps)
    const int mt = wave;
    for (int nt = 0; nt < 8; ++nt) {
      v8f acc = {};
#pragma unroll
      for (int kk = 0; kk < 6; ++kk)
        acc = WMMA_F16(load_a_frag(xs, mt * 16, kk * 32, 192),
                       load_b_frag_cm(h1wh, kk * 32, nt * 16, 192), acc);
      const int ncol = nt * 16 + (lane & 15);
      const int mbase = mt * 16 + 8 * (lane >> 4);
#pragma unroll
      for (int v = 0; v < 8; ++v)
        y1[(mbase + v) * 128 + ncol] = (_Float16)gelu_exact(acc[v] + h1_b[ncol]);
    }
  }
  __syncthreads();
  { // h2: K=128 (4 steps)
    const int mt = wave;
    for (int nt = 0; nt < 8; ++nt) {
      v8f acc = {};
#pragma unroll
      for (int kk = 0; kk < 4; ++kk)
        acc = WMMA_F16(load_a_frag(y1, mt * 16, kk * 32, 128),
                       load_b_frag_cm(h2wh, kk * 32, nt * 16, 128), acc);
      const int ncol = nt * 16 + (lane & 15);
      const int mbase = mt * 16 + 8 * (lane >> 4);
#pragma unroll
      for (int v = 0; v < 8; ++v)
        y2[(mbase + v) * 128 + ncol] = (_Float16)gelu_exact(acc[v] + h2_b[ncol]);
    }
  }
  __syncthreads();

  if (tid < 64) {                    // h3 (128->3) + finalize, one row/thread
    int r = r0 + tid;
    if (r < N) {
      float o0 = h3_b[0], o1 = h3_b[1], o2 = h3_b[2];
      for (int k = 0; k < 128; ++k) {
        float v = (float)y2[tid * 128 + k];
        o0 += v * h3_w[k];
        o1 += v * h3_w[128 + k];
        o2 += v * h3_w[256 + k];
      }
      float ls   = fminf(fmaxf(o2, -6.f), 3.f);
      float conf = 1.f / fmaxf(expf(ls), 1e-4f);
      const float* M = p2s + (size_t)r * 4;
      out[(size_t)r * 2 + 0] = M[0] * o0 + M[1] * o1;
      out[(size_t)r * 2 + 1] = M[2] * o0 + M[3] * o1;
      out[(size_t)(2 * N) + r] = o0;
      out[(size_t)(3 * N) + r] = o1;
      out[(size_t)(4 * N) + r] = ls;
      out[(size_t)(5 * N) + r] = conf;
    }
  }
}

// ---------------------------------------------------------------------------
extern "C" void kernel_launch(void* const* d_in, const int* in_sizes, int n_in,
                              void* d_out, int out_size, void* d_ws, size_t ws_size,
                              hipStream_t stream) {
  const float* bottleneck = (const float*)d_in[0];
  const float* vis        = (const float*)d_in[1];
  const float* pos        = (const float*)d_in[2];
  const float* p2s        = (const float*)d_in[3];
  const int*   fh = (const int*)d_in[4];
  const int*   fw = (const int*)d_in[5];
  const int*   vh = (const int*)d_in[6];
  const int*   vw = (const int*)d_in[7];
  const float* dw_w = (const float*)d_in[8];
  const float* dw_b = (const float*)d_in[9];
  const float* ln_g = (const float*)d_in[10];
  const float* ln_b = (const float*)d_in[11];
  const float* pw1_w = (const float*)d_in[12];
  const float* pw1_b = (const float*)d_in[13];
  const float* pw2_w = (const float*)d_in[14];
  const float* pw2_b = (const float*)d_in[15];
  const float* cn_gamma = (const float*)d_in[16];
  const float* bn1x1_w  = (const float*)d_in[17];
  const float* bn1x1_b  = (const float*)d_in[18];
  const float* st1_w = (const float*)d_in[19];
  const float* st1_b = (const float*)d_in[20];
  const float* st2_w = (const float*)d_in[21];
  const float* st2_b = (const float*)d_in[22];
  const float* h1_w  = (const float*)d_in[23];
  const float* h1_b  = (const float*)d_in[24];
  const float* h2_w  = (const float*)d_in[25];
  const float* h2_b  = (const float*)d_in[26];
  const float* h3_w  = (const float*)d_in[27];
  const float* h3_b  = (const float*)d_in[28];

  const int N = in_sizes[2] / 2;                     // 4096 sources

  _Float16* wsh = (_Float16*)d_ws;
  char*     wsb = (char*)d_ws;
  const size_t off_bnvec = (F16_TOTAL * 2 + 255) & ~(size_t)255;
  const size_t off_stvec = off_bnvec + (size_t)N * 128 * sizeof(float);
  const size_t off_stage = off_stvec + (size_t)N * 64 * sizeof(float);
  float*    bn_vec = (float*)(wsb + off_bnvec);
  float*    st_vec = (float*)(wsb + off_stvec);
  _Float16* stage  = (_Float16*)(wsb + off_stage);   // [N][289][64] f16

  prep_weights_kernel<<<(int)((F16_TOTAL + 255) / 256), 256, 0, stream>>>(
      pw1_w, pw2_w, bn1x1_w, st1_w, st2_w, h1_w, h2_w, wsh);

  bn_branch_kernel<<<N, 256, 0, stream>>>(
      bottleneck, pos, fh, fw, vh, vw, dw_w, dw_b, ln_g, ln_b,
      pw1_b, pw2_b, cn_gamma, bn1x1_b,
      wsh + OFF_W1B, wsh + OFF_W2B, wsh + OFF_BNW, bn_vec);

  stem1_kernel<<<N, 256, 0, stream>>>(vis, pos, st1_b, wsh + OFF_ST1W, stage);
  stem2_kernel<<<N, 256, 0, stream>>>(stage, st2_b, wsh + OFF_ST2W, st_vec);

  head_kernel<<<(N + 63) / 64, 128, 0, stream>>>(
      bn_vec, st_vec, p2s, h1_b, h2_b, h3_w, h3_b,
      wsh + OFF_H1W, wsh + OFF_H2W, (float*)d_out, N);
}
